// RowCaiQuantLinear_63788854280517
// MI455X (gfx1250) — compile-verified, matrix operand order checked by reading, not run on hardware
//
#include <hip/hip_runtime.h>
#include <stdint.h>

typedef _Float16 half_t;
typedef __attribute__((ext_vector_type(16))) _Float16 v16h;
typedef __attribute__((ext_vector_type(8)))  _Float16 v8h;
typedef __attribute__((ext_vector_type(4)))  _Float16 v4h;
typedef __attribute__((ext_vector_type(2)))  _Float16 h2;
typedef __attribute__((ext_vector_type(8)))  float    v8f;
typedef __attribute__((ext_vector_type(4)))  unsigned u32x4;
typedef __attribute__((ext_vector_type(4)))  int      i32x4;
typedef __attribute__((ext_vector_type(8)))  int      i32x8;

#define K_DIM 4096
#define N_DIM 4096
#define BM 256
#define BN 128
#define BK 64
#define KTILES (K_DIM / BK)     // 64
#define LDSS 72                 // 64 halves + 8 pad -> 144B row stride (16B aligned)

// ---- Tensor Data Mover: async 2D tile load Global -> LDS (padded rows) ----
__device__ __forceinline__ void tdm_load_tile_A(const half_t* gsrc, const half_t* lds_dst, int M)
{
    const uint64_t ga = (uint64_t)(uintptr_t)gsrc;
    const uint32_t la = (uint32_t)(uintptr_t)lds_dst;
    u32x4 g0;
    g0[0] = 1u;                                                   // count=1 (valid user D#)
    g0[1] = la;                                                   // lds_addr
    g0[2] = (uint32_t)ga;                                         // global_addr[31:0]
    g0[3] = (uint32_t)((ga >> 32) & 0x01FFFFFFu) | (2u << 30);    // global_addr[56:32] | type=2
    i32x8 g1;
    g1[0] = (1 << 16)        // data_size = 2 bytes
          | (1 << 20)        // pad_enable
          | (4 << 22)        // pad_interval: 32 DWORDs (=128B = one 64-half row)
          | (3 << 25);       // pad_amount: 4 DWORDs (=16B = 8 halves)
    g1[1] = (int)((K_DIM & 0xFFFF) << 16);                           // tensor_dim0 lo16
    g1[2] = (int)(((K_DIM >> 16) & 0xFFFF) | ((M & 0xFFFF) << 16));  // dim0 hi | tensor_dim1 lo
    g1[3] = (int)(((M >> 16) & 0xFFFF) | (BK << 16));                // dim1 hi | tile_dim0=64
    g1[4] = BM;                                                      // tile_dim1=256 | tile_dim2=0
    g1[5] = K_DIM;                                                   // tensor_dim0_stride lo32
    g1[6] = 0;
    g1[7] = 0;
    i32x4 gz = {0, 0, 0, 0};
#if defined(__clang_major__) && (__clang_major__ >= 23)
    i32x8 gz8 = {0, 0, 0, 0, 0, 0, 0, 0};
    __builtin_amdgcn_tensor_load_to_lds(g0, g1, gz, gz, gz8, 0);
#else
    __builtin_amdgcn_tensor_load_to_lds(g0, g1, gz, gz, 0);
#endif
}

// ---- fast GPTQ nibble dequant: fp16-magic, packed pk_fma, natural K order ----
__device__ __forceinline__ void dequant_store(half_t* Wb, i32x4 qv, int zr, v4h sv,
                                              int dr, int dc)
{
    #pragma unroll
    for (int j = 0; j < 4; ++j) {
        const unsigned q    = (unsigned)qv[j];
        const unsigned znib = ((unsigned)zr >> (((dc + j) & 7) * 4)) & 0xFu;
        // half bits of (1024 + (znib+1)) built without cvt: 0x6400 | (znib+1)
        union { unsigned short us; _Float16 h; } hb;
        hb.us = (unsigned short)(0x6401u + znib);
        const _Float16 s   = sv[j];
        const _Float16 off = (_Float16)(-(hb.h * s));     // -(1024+z)*s
        h2 s2;   s2[0] = s;    s2[1] = s;
        h2 off2; off2[0] = off; off2[1] = off;
        union { unsigned u; h2 h; } cv;
        union { v8h v; h2 h[4]; } w;
        #pragma unroll
        for (int i = 0; i < 4; ++i) {
            const unsigned raw = q >> (8 * i);            // nibbles (2i, 2i+1)
            cv.u = ((raw & 0x0000000Fu) | ((raw & 0x000000F0u) << 12)) | 0x64006400u;
            w.h[i] = cv.h * s2 + off2;                    // v_pk_fma_f16
        }
        *(v8h*)(Wb + (size_t)(dc + j) * LDSS + dr * 8) = w.v;
    }
}

__global__ __launch_bounds__(256)
void gptq_w4a16_wmma(const half_t* __restrict__ x,
                     const int*    __restrict__ qweight,
                     const int*    __restrict__ qzeros,
                     const half_t* __restrict__ scales,
                     const half_t* __restrict__ bias,
                     float*        __restrict__ out,
                     int M)
{
    __shared__ half_t As[2][BM * LDSS];   // x tile, [m][k], padded rows (72 KB)
    __shared__ half_t Ws[2][BN * LDSS];   // dequantized W tile, TRANSPOSED [n][k] (36 KB)

    const int tid  = threadIdx.x;
    const int lane = tid & 31;
    const int wid  = tid >> 5;
    const int wm   = wid & 3;             // 4 wave-rows over 256 M (64 rows each)
    const int wn   = wid >> 2;            // 2 wave-cols over 128 N (64 cols each)
    const int m0   = blockIdx.y * BM;
    const int n0   = blockIdx.x * BN;

    // dequant assignment: k-subrow dr (0..7), 4 consecutive columns at dc
    const int dr = tid >> 5;              // 0..7
    const int dc = (tid & 31) * 4;        // 0..124, step 4

    v8f acc[4][4] = {};                   // 16 x (16x16) tiles per wave = 64M x 64N

    const int lrow = lane & 15;
    const int kb   = (lane >> 4) << 3;    // 0 / 8 per ISA 16-bit fragment layout

    i32x4 qreg; int zr; v4h sreg;

    // ---- prologue: stage tile 0 into buffer 0
    qreg = *(const i32x4*)(qweight + (size_t)dr * N_DIM + n0 + dc);
    zr   = qzeros[(size_t)((n0 + dc) >> 3)];
    sreg = *(const v4h*)(scales + n0 + dc);
    if (wid == 0) tdm_load_tile_A(x + (size_t)m0 * K_DIM, &As[0][0], M);
    dequant_store(&Ws[0][0], qreg, zr, sreg, dr, dc);
    if (wid == 0) __builtin_amdgcn_s_wait_tensorcnt(0);
    __syncthreads();

    #pragma clang loop unroll(disable)
    for (int t = 0; t < KTILES; ++t) {
        const int p = t & 1;

        // ---- issue next tile's async A-load (TDM) + W fetch into registers
        const int t1 = t + 1;
        if (t1 < KTILES) {
            if (wid == 0)
                tdm_load_tile_A(x + (size_t)m0 * K_DIM + t1 * BK, &As[p ^ 1][0], M);
            qreg = *(const i32x4*)(qweight + (size_t)(t1 * 8 + dr) * N_DIM + n0 + dc);
            if ((t1 & 1) == 0) {          // group g = t1>>1 advances only on even t1
                const int g = t1 >> 1;
                zr   = qzeros[(size_t)g * (N_DIM / 8) + ((n0 + dc) >> 3)];
                sreg = *(const v4h*)(scales + (size_t)g * N_DIM + n0 + dc);
            }
        }

        // ---- compute on buffer p: 2 k-substeps of 32, 32 WMMAs per wave
        const half_t* Ab = &As[p][0];
        const half_t* Wb = &Ws[p][0];
        #pragma unroll
        for (int ks = 0; ks < 2; ++ks) {
            v16h afrag[4];
            #pragma unroll
            for (int tm = 0; tm < 4; ++tm) {
                const half_t* base = Ab + (wm * 64 + tm * 16 + lrow) * LDSS + ks * 32 + kb;
                const v8h lo = *(const v8h*)(base);        // K = kb..kb+7
                const v8h hi = *(const v8h*)(base + 16);   // K = kb+16..kb+23
                afrag[tm] = __builtin_shufflevector(lo, hi,
                    0, 1, 2, 3, 4, 5, 6, 7, 8, 9, 10, 11, 12, 13, 14, 15);
            }
            #pragma unroll
            for (int tn = 0; tn < 4; ++tn) {
                const half_t* base = Wb + (wn * 64 + tn * 16 + lrow) * LDSS + ks * 32 + kb;
                const v8h lo = *(const v8h*)(base);
                const v8h hi = *(const v8h*)(base + 16);
                const v16h bfrag = __builtin_shufflevector(lo, hi,
                    0, 1, 2, 3, 4, 5, 6, 7, 8, 9, 10, 11, 12, 13, 14, 15);
                #pragma unroll
                for (int tm = 0; tm < 4; ++tm) {
                    acc[tm][tn] = __builtin_amdgcn_wmma_f32_16x16x32_f16(
                        false, afrag[tm], false, bfrag,
                        (short)0, acc[tm][tn], false, false);
                }
            }
        }

        // ---- dequant next tile into the other buffer (overlaps with WMMA issue)
        if (t1 < KTILES)
            dequant_store(&Ws[p ^ 1][0], qreg, zr, sreg, dr, dc);

        if (wid == 0) __builtin_amdgcn_s_wait_tensorcnt(0);
        __syncthreads();
    }

    // ---- epilogue: C/D layout (VGPR e -> M = e + 8*(lane>=16), N = lane&15)
    const int mhalf = (lane >> 4) << 3;
    #pragma unroll
    for (int tm = 0; tm < 4; ++tm) {
        #pragma unroll
        for (int tn = 0; tn < 4; ++tn) {
            const int n = n0 + wn * 64 + tn * 16 + lrow;
            const half_t bv = bias[n];
            #pragma unroll
            for (int e = 0; e < 8; ++e) {
                const int m = m0 + wm * 64 + tm * 16 + mhalf + e;
                const half_t v = (half_t)acc[tm][tn][e] + bv;  // fp16 round + fp16 bias
                out[(size_t)m * N_DIM + n] = (float)v;
            }
        }
    }
}

extern "C" void kernel_launch(void* const* d_in, const int* in_sizes, int n_in,
                              void* d_out, int out_size, void* d_ws, size_t ws_size,
                              hipStream_t stream) {
    const half_t* x       = (const half_t*)d_in[0];
    const int*    qweight = (const int*)d_in[1];
    const int*    qzeros  = (const int*)d_in[2];
    const half_t* scales  = (const half_t*)d_in[3];
    const half_t* bias    = (const half_t*)d_in[4];
    float*        out     = (float*)d_out;

    const int M = in_sizes[0] / K_DIM;    // 8192
    dim3 grid(N_DIM / BN, M / BM);        // 32 x 32
    dim3 block(256);
    gptq_w4a16_wmma<<<grid, block, 0, stream>>>(x, qweight, qzeros, scales, bias, out, M);
}
